// FP8Linear_22256520528510
// MI455X (gfx1250) — 8-bit matrix operand order repaired and checked by reading, compiled, not run on hardware
//
#include <hip/hip_runtime.h>
#include <stdint.h>


typedef int   v16i __attribute__((ext_vector_type(16)));
typedef float v8f  __attribute__((ext_vector_type(8)));
typedef int   v4i_vs __attribute__((vector_size(16)));

#define M_TOT 8192
#define K_TOT 4096
#define N_TOT 11008
#define KB    32
#define FP8_MAX 448.0f

#define LDS_PITCH 144
#define A_BYTES   (128 * LDS_PITCH)
#define BUF_BYTES (2 * A_BYTES)
#define AS1 __attribute__((address_space(1)))
#define AS3 __attribute__((address_space(3)))

__device__ __forceinline__ void async_copy_b128(const void* gptr, void* lptr)
{
#if defined(__has_builtin) && __has_builtin(__builtin_amdgcn_global_load_async_to_lds_b128)
    __builtin_amdgcn_global_load_async_to_lds_b128(
        (AS1 v4i_vs*)(uintptr_t)gptr,
        (AS3 v4i_vs*)(uint32_t)(uintptr_t)lptr, 0, 0);
#else
    asm volatile("global_load_async_to_lds_b128 %0, %1, off"
                 :: "v"((uint32_t)(uintptr_t)lptr), "v"(gptr)
                 : "memory");
#endif
}

__device__ __forceinline__ void wait_async0()
{
#if defined(__has_builtin) && __has_builtin(__builtin_amdgcn_s_wait_asynccnt)
    __builtin_amdgcn_s_wait_asynccnt(0);
#else
    asm volatile("s_wait_asynccnt 0x0" ::: "memory");
#endif
}

__global__ __launch_bounds__(256)
void quant_x_kernel(const float* __restrict__ x, uint8_t* __restrict__ xq,
                    float* __restrict__ asT)
{
    const int tid  = threadIdx.x;
    const int lane = tid & 31;
    const int wave = tid >> 5;
    const int gid  = blockIdx.x * 8 + wave;
    const int m    = gid >> 5;
    const int kb   = gid & 31;

    const float4* src = (const float4*)(x + (size_t)m * K_TOT + kb * 128);
    const float4 v = src[lane];

    float amax = fmaxf(fmaxf(fabsf(v.x), fabsf(v.y)),
                       fmaxf(fabsf(v.z), fabsf(v.w)));
#pragma unroll
    for (int off = 16; off > 0; off >>= 1)
        amax = fmaxf(amax, __shfl_xor(amax, off, 32));

    const float s   = amax * (1.0f / FP8_MAX);
    const float inv = (amax > 0.0f) ? (FP8_MAX / amax) : 0.0f;

    int q = __builtin_amdgcn_cvt_pk_fp8_f32(v.x * inv, v.y * inv, 0, false);
    q     = __builtin_amdgcn_cvt_pk_fp8_f32(v.z * inv, v.w * inv, q, true);

    ((int*)(xq + (size_t)m * K_TOT + kb * 128))[lane] = q;
    if (lane == 0) asT[(size_t)kb * M_TOT + m] = s;
}

__global__ __launch_bounds__(256)
void quant_w_kernel(const float* __restrict__ w, uint8_t* __restrict__ wq)
{
    const size_t i = (size_t)blockIdx.x * 256 + threadIdx.x;
    if (i >= (size_t)N_TOT * K_TOT / 4) return;
    const float4 v = ((const float4*)w)[i];
    int q = __builtin_amdgcn_cvt_pk_fp8_f32(v.x, v.y, 0, false);
    q     = __builtin_amdgcn_cvt_pk_fp8_f32(v.z, v.w, q, true);
    ((int*)wq)[i] = q;
}

__device__ __forceinline__ void stage_kblock(const uint8_t* __restrict__ gA,
                                             const uint8_t* __restrict__ gB,
                                             uint8_t* lA, uint8_t* lB, int t)
{
#pragma unroll
    for (int i = 0; i < 4; ++i) {
        const int j   = i * 256 + t;
        const int row = j >> 3;
        const int co  = (j & 7) * 16;
        async_copy_b128(gA + (size_t)row * K_TOT + co, lA + row * LDS_PITCH + co);
    }
#pragma unroll
    for (int i = 0; i < 4; ++i) {
        const int j   = i * 256 + t;
        const int row = j >> 3;
        const int co  = (j & 7) * 16;
        async_copy_b128(gB + (size_t)row * K_TOT + co, lB + row * LDS_PITCH + co);
    }
}

__global__ __launch_bounds__(256)
void gemm_fp8_kernel(const uint8_t* __restrict__ xq,
                     const float*   __restrict__ asT,
                     const uint8_t* __restrict__ wq,
                     const float*   __restrict__ wscale,
                     const float*   __restrict__ bias,
                     float*         __restrict__ out)
{
    __shared__ __align__(16) uint8_t smem[2 * BUF_BYTES];

    const int tid  = threadIdx.x;
    const int lane = tid & 31;
    const int wave = tid >> 5;
    const int half = lane >> 4;
    const int l15  = lane & 15;

    const int bm = blockIdx.x % (M_TOT / 128);
    const int bn = blockIdx.x / (M_TOT / 128);

    const int mwave = (wave & 1) * 64;
    const int nwave = (wave >> 1) * 32;
    const int mbase = bm * 128 + mwave;
    const int nbase = bn * 128 + nwave;

    const uint8_t* gA0 = xq + (size_t)bm * 128 * K_TOT;
    const uint8_t* gB0 = wq + (size_t)bn * 128 * K_TOT;

    v8f acc[4][2];
#pragma unroll
    for (int mi = 0; mi < 4; ++mi)
#pragma unroll
        for (int ni = 0; ni < 2; ++ni)
            acc[mi][ni] = (v8f){};

    stage_kblock(gA0, gB0, smem, smem + A_BYTES, tid);

    for (int kb = 0; kb < KB; ++kb) {
        wait_async0();
        __syncthreads();

        const int cur = kb & 1;
        if (kb + 1 < KB)
            stage_kblock(gA0 + (kb + 1) * 128, gB0 + (kb + 1) * 128,
                         smem + (cur ^ 1) * BUF_BYTES,
                         smem + (cur ^ 1) * BUF_BYTES + A_BYTES, tid);

        const uint8_t* Ab = smem + cur * BUF_BYTES;
        const uint8_t* Bb = Ab + A_BYTES;

        const float ws = wscale[bn * KB + kb];

        v16i B[2];
#pragma unroll
        for (int ni = 0; ni < 2; ++ni) {
            const uint8_t* bp = Bb + (nwave + ni * 16 + l15) * LDS_PITCH + half * 16;
#pragma unroll
            for (int g = 0; g < 4; ++g) {
                const int4 d = *(const int4*)(bp + g * 32);
                B[ni][4 * g + 0] = d.x;  B[ni][4 * g + 1] = d.y;
                B[ni][4 * g + 2] = d.z;  B[ni][4 * g + 3] = d.w;
            }
        }

#pragma unroll
        for (int mi = 0; mi < 4; ++mi) {
            const uint8_t* ap = Ab + (mwave + mi * 16 + l15) * LDS_PITCH + half * 16;
            v16i A;
#pragma unroll
            for (int c = 0; c < 8; ++c) {
                const int2 d = *(const int2*)(ap + 32 * (c >> 1) + 8 * (c & 1));
                A[2 * c + 0] = d.x;
                A[2 * c + 1] = d.y;
            }

            const float* sp = asT + (size_t)kb * M_TOT + mbase + mi * 16 + half * 8;
            const float4 s0 = ((const float4*)sp)[0];
            const float4 s1 = ((const float4*)sp)[1];
            const float srow[8] = { s0.x, s0.y, s0.z, s0.w,
                                    s1.x, s1.y, s1.z, s1.w };

#pragma unroll
            for (int ni = 0; ni < 2; ++ni) {
                const v8f z = (v8f){};
                const v8f p = __builtin_amdgcn_wmma_f32_16x16x128_fp8_fp8(
                    A, B[ni], (short)0, z, false, false);
#pragma unroll
                for (int r = 0; r < 8; ++r)
                    acc[mi][ni][r] += p[r] * (srow[r] * ws);
            }
        }
    }

#pragma unroll
    for (int ni = 0; ni < 2; ++ni) {
        const float b = bias[nbase + ni * 16 + l15];
#pragma unroll
        for (int mi = 0; mi < 4; ++mi)
#pragma unroll
            for (int r = 0; r < 8; ++r)
                out[(size_t)(mbase + mi * 16 + half * 8 + r) * N_TOT
                    + nbase + ni * 16 + l15] = acc[mi][ni][r] + b;
    }
}

extern "C" void kernel_launch(void* const* d_in, const int* in_sizes, int n_in,
                              void* d_out, int out_size, void* d_ws, size_t ws_size,
                              hipStream_t stream)
{
    const float* x      = (const float*)d_in[0];
    const float* weight = (const float*)d_in[1];
    const float* wscale = (const float*)d_in[2];
    const float* bias   = (const float*)d_in[3];
    float* out = (float*)d_out;

    uint8_t* ws8 = (uint8_t*)d_ws;
    uint8_t* xq  = ws8;
    uint8_t* wq  = ws8 + (size_t)M_TOT * K_TOT;
    float*   asT = (float*)(ws8 + (size_t)M_TOT * K_TOT + (size_t)N_TOT * K_TOT);

    quant_x_kernel<<<(M_TOT * KB) / 8, 256, 0, stream>>>(x, xq, asT);

    const size_t wdwords = (size_t)N_TOT * K_TOT / 4;
    quant_w_kernel<<<(unsigned)((wdwords + 255) / 256), 256, 0, stream>>>(weight, wq);

    gemm_fp8_kernel<<<(M_TOT / 128) * (N_TOT / 128), 256, 0, stream>>>(
        xq, asT, wq, wscale, bias, out);
}
